// FastFeedForwardNetwork_78872779424238
// MI455X (gfx1250) — matrix operand order repaired and checked by reading, not compiled, not run on hardware
//
#include <hip/hip_runtime.h>
#include <stdint.h>

// Problem constants (from reference): B=4, T=2048, D=1024, L=16, K=128, O=1024
#define NTOK   8192
#define DIN    1024
#define KHALF  128
#define TWOK   256
#define OUTD   1024
#define NLEAF  16
#define MTILE  32                 // tokens per block (2 WMMA row-tiles)
#define MAXTILE (NTOK / MTILE)    // 256 tiles worst case

typedef __attribute__((ext_vector_type(16))) __bf16       v16bf;
typedef __attribute__((ext_vector_type(8)))  float        v8f;
typedef __attribute__((ext_vector_type(4)))  unsigned int u32x4;

struct Frag32 { u32x4 lo, hi; };

__device__ __forceinline__ unsigned short f2bf(float f) {
  unsigned u = __builtin_bit_cast(unsigned, f);
  u += 0x7FFFu + ((u >> 16) & 1u);          // round-to-nearest-even
  return (unsigned short)(u >> 16);
}
__device__ __forceinline__ unsigned pack2bf(float a, float b) {
  return (unsigned)f2bf(a) | ((unsigned)f2bf(b) << 16);
}
__device__ __forceinline__ v16bf frag_lds(const unsigned short* p) {
  Frag32 f;
  f.lo = *(const u32x4*)(p);        // halfs [0..7]
  f.hi = *(const u32x4*)(p + 8);    // halfs [8..15]
  return __builtin_bit_cast(v16bf, f);
}
__device__ __forceinline__ v16bf frag_glb(const unsigned* p) {
  Frag32 f;
  f.lo = *(const u32x4*)(p);
  f.hi = *(const u32x4*)(p + 4);
  return __builtin_bit_cast(v16bf, f);
}
#define WMMA_BF16(A, B, C) \
  __builtin_amdgcn_wmma_f32_16x16x32_bf16(false, (A), false, (B), (short)0, (C), false, false)

__global__ void ffn_init(int* counts) {
  if (threadIdx.x < NLEAF) counts[threadIdx.x] = 0;
}

// Swizzle w1 f32 [L][1024][256] -> bf16 WMMA B-fragment layout.
// dword idx = ((((leaf*32 + kb)*16 + nb)*32) + lane)*8 + j
// lane<16: N = nb*16+lane,     K = kb*32 + {2j,2j+1}
// lane>=16: N = nb*16+lane-16, K = kb*32 + 16 + {2j,2j+1}
__global__ void ffn_prep_w1(const float* __restrict__ w1, unsigned* __restrict__ w1s) {
  unsigned idx  = blockIdx.x * 256u + threadIdx.x;   // 0 .. 2M-1
  unsigned j    =  idx        & 7u;
  unsigned lane = (idx >> 3)  & 31u;
  unsigned nb   = (idx >> 8)  & 15u;
  unsigned kb   = (idx >> 12) & 31u;
  unsigned leaf =  idx >> 17;
  unsigned n  = nb * 16u + (lane & 15u);
  unsigned k0 = kb * 32u + ((lane & 16u) ? 16u : 0u) + 2u * j;
  const float* src = w1 + (size_t)leaf * (DIN * TWOK);
  w1s[idx] = pack2bf(src[k0 * TWOK + n], src[(k0 + 1) * TWOK + n]);
}

// Swizzle w2 f32 [L][128][1024] -> bf16 WMMA B-fragment layout.
__global__ void ffn_prep_w2(const float* __restrict__ w2, unsigned* __restrict__ w2s) {
  unsigned idx  = blockIdx.x * 256u + threadIdx.x;   // 0 .. 1M-1
  unsigned j    =  idx        & 7u;
  unsigned lane = (idx >> 3)  & 31u;
  unsigned nb   = (idx >> 8)  & 63u;
  unsigned kb   = (idx >> 14) & 3u;
  unsigned leaf =  idx >> 16;
  unsigned n  = nb * 16u + (lane & 15u);
  unsigned k0 = kb * 32u + ((lane & 16u) ? 16u : 0u) + 2u * j;
  const float* src = w2 + (size_t)leaf * (KHALF * OUTD);
  w2s[idx] = pack2bf(src[k0 * OUTD + n], src[(k0 + 1) * OUTD + n]);
}

// Hard routing: leaf bit m = (logit_m > 0 ? 0 : 1), MSB = level 0.
// Wave per token, f32 dot products (exact sign decisions), atomic bucketing.
__global__ void ffn_router(const float* __restrict__ x,
                           const float* __restrict__ wnode,
                           const float* __restrict__ bnode,
                           int* __restrict__ counts,
                           int* __restrict__ lists) {
  int wave = threadIdx.x >> 5;
  int lane = threadIdx.x & 31;
  int token = blockIdx.x * 8 + wave;
  const float* xr = x + (size_t)token * DIN;
  float a0 = 0.f, a1 = 0.f, a2 = 0.f, a3 = 0.f;
  for (int d = lane; d < DIN; d += 32) {
    float xv = xr[d];
    a0 += xv * wnode[d];
    a1 += xv * wnode[DIN + d];
    a2 += xv * wnode[2 * DIN + d];
    a3 += xv * wnode[3 * DIN + d];
  }
  for (int off = 16; off >= 1; off >>= 1) {
    a0 += __shfl_xor(a0, off, 32);
    a1 += __shfl_xor(a1, off, 32);
    a2 += __shfl_xor(a2, off, 32);
    a3 += __shfl_xor(a3, off, 32);
  }
  if (lane == 0) {
    float l0 = a0 + bnode[0], l1 = a1 + bnode[1];
    float l2 = a2 + bnode[2], l3 = a3 + bnode[3];
    int leaf = ((l0 > 0.f) ? 0 : 8) | ((l1 > 0.f) ? 0 : 4) |
               ((l2 > 0.f) ? 0 : 2) | ((l3 > 0.f) ? 0 : 1);
    int pos = atomicAdd(&counts[leaf], 1);
    lists[leaf * NTOK + pos] = token;
  }
}

#define XROW 1032   // 1024 + 8 halfs pad (keeps 16B alignment, spreads banks)
#define HROW 260
#define AROW 136    // 128 + 8 halfs pad

__global__ void __launch_bounds__(512)
ffn_leaf(const float* __restrict__ x,
         const float* __restrict__ b1,
         const float* __restrict__ b2,
         const int* __restrict__ counts,
         const int* __restrict__ lists,
         const unsigned* __restrict__ w1s,
         const unsigned* __restrict__ w2s,
         float* __restrict__ out) {
  __shared__ unsigned short Xbf[MTILE * XROW];   // gathered inputs, bf16 (~66 KB)
  __shared__ float          Hs [MTILE * HROW];   // hidden pre-GLU, f32  (~33 KB)
  __shared__ unsigned short Abf[MTILE * AROW];   // GLU output, bf16     (~9 KB)
  __shared__ int            toks[MTILE];

  const int leaf = blockIdx.x;
  const int tile = blockIdx.y;
  const int nTok = counts[leaf];
  if (tile * MTILE >= nTok) return;              // wave-uniform early exit

  const int tid  = threadIdx.x;
  const int wave = tid >> 5;
  const int lane = tid & 31;

  if (tid < MTILE) {
    int p = tile * MTILE + tid;
    toks[tid] = (p < nTok) ? lists[leaf * NTOK + p] : -1;
  }
  __syncthreads();

  // Gather 32 token rows into LDS as bf16 (coalesced over columns)
  for (int i = tid; i < MTILE * DIN; i += 512) {
    int r = i >> 10, c = i & (DIN - 1);
    int tok = toks[r];
    float v = (tok >= 0) ? x[(size_t)tok * DIN + c] : 0.f;
    Xbf[r * XROW + c] = f2bf(v);
  }
  __syncthreads();

  const int hiSel = (lane & 16) ? 16 : 0;     // A-frag K-chunk select
  const int rbase = (lane & 16) ? 8 : 0;      // C-tile row group
  const int arow0 = (lane & 15);              // M-tile 0 fragment row
  const int arow1 = arow0 + 16;               // M-tile 1 fragment row

  // ---- stage A: H(32x256) = Xbf(32x1024) @ W1[leaf](1024x256)
  // wave -> one 16-col strip, both 16-row tiles; B fragment reused across M-tiles.
  v8f h0 = {}, h1 = {};
  const unsigned* w1p = w1s + (((unsigned)leaf * 32u * 16u + wave) * 256u) + lane * 8u;
  for (int kb = 0; kb < 32; ++kb) {
    v16bf bfrag = frag_glb(w1p + kb * (16u * 256u));
    v16bf a0 = frag_lds(&Xbf[arow0 * XROW + kb * 32 + hiSel]);
    v16bf a1 = frag_lds(&Xbf[arow1 * XROW + kb * 32 + hiSel]);
    h0 = WMMA_BF16(a0, bfrag, h0);
    h1 = WMMA_BF16(a1, bfrag, h1);
  }
  {
    int col  = wave * 16 + (lane & 15);
    float bb = b1[leaf * TWOK + col];
    #pragma unroll
    for (int i = 0; i < 8; ++i) {
      Hs[(rbase + i) * HROW + col]      = h0[i] + bb;
      Hs[(16 + rbase + i) * HROW + col] = h1[i] + bb;
    }
  }
  __syncthreads();

  // GLU: a = h[:,128:] * h[:,:128], to bf16 A-operand
  for (int i = tid; i < MTILE * KHALF; i += 512) {
    int r = i >> 7, c = i & (KHALF - 1);
    Abf[r * AROW + c] = f2bf(Hs[r * HROW + c + KHALF] * Hs[r * HROW + c]);
  }
  __syncthreads();

  // ---- stage B: OUT(32x1024) = Abf(32x128) @ W2[leaf](128x1024)
  // wave -> 4 N-strips x 2 M-tiles; each B fragment reused across M-tiles.
  v8f o0[4] = {{}, {}, {}, {}};
  v8f o1[4] = {{}, {}, {}, {}};
  const unsigned* w2p = w2s + ((unsigned)leaf * 4u * 64u * 256u) + lane * 8u;
  #pragma unroll
  for (int kb = 0; kb < 4; ++kb) {
    v16bf a0 = frag_lds(&Abf[arow0 * AROW + kb * 32 + hiSel]);
    v16bf a1 = frag_lds(&Abf[arow1 * AROW + kb * 32 + hiSel]);
    const unsigned* wp = w2p + (unsigned)kb * (64u * 256u);
    #pragma unroll
    for (int s = 0; s < 4; ++s) {
      v16bf bfrag = frag_glb(wp + (unsigned)(wave + 16 * s) * 256u);
      o0[s] = WMMA_BF16(a0, bfrag, o0[s]);
      o1[s] = WMMA_BF16(a1, bfrag, o1[s]);
    }
  }
  {
    int   cs[4];
    float gs[4];
    #pragma unroll
    for (int s = 0; s < 4; ++s) {
      cs[s] = (wave + 16 * s) * 16 + (lane & 15);
      gs[s] = b2[leaf * OUTD + cs[s]];
    }
    #pragma unroll
    for (int i = 0; i < 8; ++i) {
      int tok0 = toks[rbase + i];
      if (tok0 >= 0) {
        float* op = out + (size_t)tok0 * OUTD;
        #pragma unroll
        for (int s = 0; s < 4; ++s) op[cs[s]] = o0[s][i] + gs[s];
      }
      int tok1 = toks[16 + rbase + i];
      if (tok1 >= 0) {
        float* op = out + (size_t)tok1 * OUTD;
        #pragma unroll
        for (int s = 0; s < 4; ++s) op[cs[s]] = o1[s][i] + gs[s];
      }
    }
  }
}

extern "C" void kernel_launch(void* const* d_in, const int* in_sizes, int n_in,
                              void* d_out, int out_size, void* d_ws, size_t ws_size,
                              hipStream_t stream) {
  (void)in_sizes; (void)n_in; (void)out_size; (void)ws_size;
  const float* x     = (const float*)d_in[0];
  const float* w1    = (const float*)d_in[1];
  const float* w2    = (const float*)d_in[2];
  const float* b1    = (const float*)d_in[3];
  const float* b2    = (const float*)d_in[4];
  const float* wnode = (const float*)d_in[5];
  const float* bnode = (const float*)d_in[6];
  float* out = (float*)d_out;

  char* ws = (char*)d_ws;
  int*      counts = (int*)ws;                       // 16 ints
  int*      lists  = (int*)(ws + 256);               // 16*8192 ints = 512 KB
  unsigned* w1s    = (unsigned*)(ws + (1u << 20));   // 8 MB bf16 swizzled w1
  unsigned* w2s    = (unsigned*)(ws + (9u << 20));   // 4 MB bf16 swizzled w2

  ffn_init   <<<1,                    32, 0, stream>>>(counts);
  ffn_prep_w1<<<8192,                256, 0, stream>>>(w1, w1s);
  ffn_prep_w2<<<4096,                256, 0, stream>>>(w2, w2s);
  ffn_router <<<NTOK / 8,            256, 0, stream>>>(x, wnode, bnode, counts, lists);
  ffn_leaf   <<<dim3(NLEAF, MAXTILE), 512, 0, stream>>>(x, b1, b2, counts, lists,
                                                        w1s, w2s, out);
}
